// GraphModelSageMeanMax_40948218200431
// MI455X (gfx1250) — matrix operand order repaired and checked by reading, not compiled, not run on hardware
//
#include <hip/hip_runtime.h>
#include <hip/hip_bf16.h>

typedef __attribute__((ext_vector_type(16))) _Float16 v16h;
typedef __attribute__((ext_vector_type(8)))  float    v8f;

#define IN_C 128
#define HID  128
#define OUT_C 64
#define NGR  128

// ---------------------------------------------------------------- utility
__global__ void zero_f32(float* __restrict__ p, int n) {
    int i = blockIdx.x * blockDim.x + threadIdx.x;
    if (i < n) p[i] = 0.0f;
}

// ---------------------------------------------------------------- weight prep
// Swizzle a K x Ncols f32 weight (possibly concatenation of wl[0:128) and
// wr[128:K)) into the WMMA B-matrix lane layout as f16.
// Layout: tile (nt, kt) -> 512 halves; lane L holds column n = nt*16 + (L&15),
// klocal = (L>>4)*8 + (idx&7) + (idx>>3)*16 (idx = 0..15, table order).
__global__ void prep_weights(const float* __restrict__ wl,
                             const float* __restrict__ wr,
                             _Float16* __restrict__ outp,
                             int Ncols, int Ktot) {
    int t = blockIdx.x * blockDim.x + threadIdx.x;
    int ktiles = Ktot >> 5;
    int total = (Ncols >> 4) * ktiles * 512;
    if (t >= total) return;
    int block  = t >> 9;
    int within = t & 511;
    int lane = within >> 4;
    int idx  = within & 15;
    int nt = block / ktiles;
    int kt = block % ktiles;
    int klocal = ((lane >> 4) << 3) + (idx & 7) + ((idx >> 3) << 4);
    int k = (kt << 5) + klocal;
    int n = (nt << 4) + (lane & 15);
    float v = (k < 128) ? wl[k * Ncols + n] : wr[(k - 128) * Ncols + n];
    outp[t] = (_Float16)v;
}

// ---------------------------------------------------------------- graph ops
__global__ void deg_kernel(const long long* __restrict__ dst,
                           float* __restrict__ deg, int nEdges) {
    int e = blockIdx.x * blockDim.x + threadIdx.x;
    if (e < nEdges) atomicAdd(&deg[(int)dst[e]], 1.0f);
}

__global__ void scatter_feats(const float* __restrict__ feats,
                              const long long* __restrict__ src,
                              const long long* __restrict__ dst,
                              float* __restrict__ summed, int nEdges) {
    int gid = blockIdx.x * blockDim.x + threadIdx.x;
    int e = gid >> 5;
    if (e >= nEdges) return;
    int lane = gid & 31;
    long long s = src[e], d = dst[e];
    const float4 v = *(const float4*)(feats + s * IN_C + lane * 4);
    float* o = summed + d * IN_C + lane * 4;
    atomicAdd(o + 0, v.x);
    atomicAdd(o + 1, v.y);
    atomicAdd(o + 2, v.z);
    atomicAdd(o + 3, v.w);
}

__global__ void normalize_kernel(float* __restrict__ summed,
                                 const float* __restrict__ deg, int n) {
    int i = blockIdx.x * blockDim.x + threadIdx.x;
    if (i < n) {
        float d = deg[i >> 7];
        summed[i] = summed[i] / (d > 1.0f ? d : 1.0f);
    }
}

// ---------------------------------------------------------------- WMMA GEMM
// h[m, 0:128] = act( [A0(m,:) | A1(m,:)] @ Bp + bias )   (K = 256, f16 WMMA)
// One wave -> 16x16 tile; 8 waves per block cover 128 output cols.
__global__ __launch_bounds__(256) void gemm_layer(
        const float* __restrict__ A0,   // aggr [n,128]
        const float* __restrict__ A1,   // x or h1 [n,128]
        const _Float16* __restrict__ Bp,// swizzled [8 ntiles][8 ktiles][512]
        const float* __restrict__ bias, // [128]
        float* __restrict__ out,        // [n,128]
        int nrows) {
    int wave = threadIdx.x >> 5;        // n-tile
    int lane = threadIdx.x & 31;
    int base = blockIdx.x << 4;         // 16 rows per block
    int rowA = base + (lane & 15);
    if (rowA >= nrows) rowA = nrows - 1;
    int khalf = lane >> 4;              // K-half select for A layout

    __builtin_prefetch(A1 + (size_t)rowA * IN_C, 0, 0);

    v8f c = {};
    const _Float16* bp = Bp + ((size_t)wave * 8) * 512 + lane * 16;
#pragma unroll
    for (int kt = 0; kt < 8; ++kt) {
        const float* srcA = (kt < 4) ? A0 : A1;
        int kk = (kt & 3) << 5;
        const float* ap = srcA + (size_t)rowA * IN_C + kk + khalf * 8;
        v16h a;
#pragma unroll
        for (int i = 0; i < 8; ++i) {
            a[i]     = (_Float16)ap[i];          // k = kk + khalf*8 + i
            a[i + 8] = (_Float16)ap[16 + i];     // k = kk + 16 + khalf*8 + i
        }
        v16h b = *(const v16h*)(bp + kt * 512);
        c = __builtin_amdgcn_wmma_f32_16x16x32_f16(false, a, false, b,
                                                   (short)0, c, false, false);
    }

    int ncol = (wave << 4) + (lane & 15);
    float bv = bias[ncol];
    int mtop = (lane >> 4) << 3;        // 0 or 8
#pragma unroll
    for (int r = 0; r < 8; ++r) {
        int m = base + mtop + r;
        if (m < nrows) {
            float v = c[r] + bv;
            out[(size_t)m * HID + ncol] = v > 0.0f ? v : 0.0f;  // ReLU
        }
    }
}

// out[m, 0:64] = gcat(m, 0:256) @ Blin + blin   (no relu), 128 rows
__global__ __launch_bounds__(128) void gemm_final(
        const float* __restrict__ A,    // gcat [128,256]
        const _Float16* __restrict__ Bp,// swizzled [4 ntiles][8 ktiles][512]
        const float* __restrict__ bias, // [64]
        float* __restrict__ out) {      // [128,64]
    int wave = threadIdx.x >> 5;        // 0..3
    int lane = threadIdx.x & 31;
    int base = blockIdx.x << 4;         // 8 blocks x 16 rows
    int rowA = base + (lane & 15);
    int khalf = lane >> 4;

    v8f c = {};
    const _Float16* bp = Bp + ((size_t)wave * 8) * 512 + lane * 16;
#pragma unroll
    for (int kt = 0; kt < 8; ++kt) {
        const float* ap = A + (size_t)rowA * 256 + (kt << 5) + khalf * 8;
        v16h a;
#pragma unroll
        for (int i = 0; i < 8; ++i) {
            a[i]     = (_Float16)ap[i];
            a[i + 8] = (_Float16)ap[16 + i];
        }
        v16h b = *(const v16h*)(bp + kt * 512);
        c = __builtin_amdgcn_wmma_f32_16x16x32_f16(false, a, false, b,
                                                   (short)0, c, false, false);
    }

    int ncol = (wave << 4) + (lane & 15);
    float bv = bias[ncol];
    int mtop = (lane >> 4) << 3;
#pragma unroll
    for (int r = 0; r < 8; ++r) {
        int m = base + mtop + r;
        out[(size_t)m * OUT_C + ncol] = c[r] + bv;
    }
}

// ---------------------------------------------------------------- pooling
__global__ void pool_kernel(const float* __restrict__ h,
                            const long long* __restrict__ batch,
                            float* __restrict__ gsum,
                            int* __restrict__ gmax,
                            float* __restrict__ cnt, int nNodes) {
    int gid = blockIdx.x * blockDim.x + threadIdx.x;
    int node = gid >> 5;
    if (node >= nNodes) return;
    int lane = gid & 31;
    int g = (int)batch[node];
    const float4 v = *(const float4*)(h + (size_t)node * HID + lane * 4);
    float* so = gsum + g * HID + lane * 4;
    int*   mo = gmax + g * HID + lane * 4;
    atomicAdd(so + 0, v.x);
    atomicAdd(so + 1, v.y);
    atomicAdd(so + 2, v.z);
    atomicAdd(so + 3, v.w);
    // h >= 0 post-ReLU: non-negative IEEE floats order like ints
    atomicMax(mo + 0, __float_as_int(v.x));
    atomicMax(mo + 1, __float_as_int(v.y));
    atomicMax(mo + 2, __float_as_int(v.z));
    atomicMax(mo + 3, __float_as_int(v.w));
    if (lane == 0) atomicAdd(&cnt[g], 1.0f);
}

__global__ void pool_finalize(const float* __restrict__ gsum,
                              const int* __restrict__ gmax,
                              const float* __restrict__ cnt,
                              float* __restrict__ gcat) {
    int i = blockIdx.x * blockDim.x + threadIdx.x;
    if (i >= NGR * 256) return;
    int g = i >> 8;
    int j = i & 255;
    float v;
    if (j < HID) {
        float cc = cnt[g];
        v = gsum[g * HID + j] / (cc > 1.0f ? cc : 1.0f);
    } else {
        v = __int_as_float(gmax[g * HID + (j - HID)]);
    }
    gcat[i] = v;
}

// ---------------------------------------------------------------- launcher
extern "C" void kernel_launch(void* const* d_in, const int* in_sizes, int n_in,
                              void* d_out, int out_size, void* d_ws, size_t ws_size,
                              hipStream_t stream) {
    const float*     x     = (const float*)d_in[0];
    const long long* ei    = (const long long*)d_in[1];
    const long long* batch = (const long long*)d_in[2];
    const float*     w1l   = (const float*)d_in[3];
    const float*     b1    = (const float*)d_in[4];
    const float*     w1r   = (const float*)d_in[5];
    const float*     w2l   = (const float*)d_in[6];
    const float*     b2    = (const float*)d_in[7];
    const float*     w2r   = (const float*)d_in[8];
    const float*     wlin  = (const float*)d_in[9];
    const float*     blin  = (const float*)d_in[10];
    float* out = (float*)d_out;

    const int nNodes = in_sizes[0] / IN_C;   // 50000
    const int nEdges = in_sizes[1] / 2;      // 800000
    const long long* src = ei;
    const long long* dst = ei + nEdges;

    // carve workspace (256B aligned blocks)
    char* ws = (char*)d_ws;
    size_t off = 0;
    auto carve = [&](size_t bytes) -> void* {
        off = (off + 255) & ~(size_t)255;
        void* p = ws + off;
        off += bytes;
        return p;
    };
    float*    deg    = (float*)carve((size_t)nNodes * sizeof(float));
    float*    summed = (float*)carve((size_t)nNodes * IN_C * sizeof(float));
    float*    h1     = (float*)carve((size_t)nNodes * HID * sizeof(float));
    float*    h2     = (float*)carve((size_t)nNodes * HID * sizeof(float));
    float*    gsum   = (float*)carve((size_t)NGR * HID * sizeof(float));
    int*      gmax   = (int*)  carve((size_t)NGR * HID * sizeof(int));
    float*    cnt    = (float*)carve((size_t)NGR * sizeof(float));
    float*    gcat   = (float*)carve((size_t)NGR * 256 * sizeof(float));
    _Float16* B1     = (_Float16*)carve((size_t)256 * HID * sizeof(_Float16));
    _Float16* B2     = (_Float16*)carve((size_t)256 * HID * sizeof(_Float16));
    _Float16* BL     = (_Float16*)carve((size_t)256 * OUT_C * sizeof(_Float16));
    if (off > ws_size) return;  // insufficient scratch; bail safely

    const int TB = 256;
    auto nb = [](long long n, int b) { return (int)((n + b - 1) / b); };

    // 0) zero accumulators
    zero_f32<<<nb(nNodes, TB), TB, 0, stream>>>(deg, nNodes);
    zero_f32<<<nb((long long)nNodes * IN_C, TB), TB, 0, stream>>>(summed, nNodes * IN_C);
    zero_f32<<<nb(NGR * HID, TB), TB, 0, stream>>>(gsum, NGR * HID);
    zero_f32<<<nb(NGR * HID, TB), TB, 0, stream>>>((float*)gmax, NGR * HID); // 0.0f bits == 0
    zero_f32<<<nb(NGR, TB), TB, 0, stream>>>(cnt, NGR);

    // 1) weight prep (f32 -> f16, WMMA B-swizzle)
    prep_weights<<<nb(8 * 8 * 512, TB), TB, 0, stream>>>(w1l, w1r, B1, HID, 256);
    prep_weights<<<nb(8 * 8 * 512, TB), TB, 0, stream>>>(w2l, w2r, B2, HID, 256);
    prep_weights<<<nb(4 * 8 * 512, TB), TB, 0, stream>>>(wlin, wlin + 128 * OUT_C, BL, OUT_C, 256);

    // 2) degrees (shared by both layers)
    deg_kernel<<<nb(nEdges, TB), TB, 0, stream>>>(dst, deg, nEdges);

    // 3) layer 1: scatter-mean + fused dual GEMM + ReLU
    scatter_feats<<<nb((long long)nEdges * 32, TB), TB, 0, stream>>>(x, src, dst, summed, nEdges);
    normalize_kernel<<<nb((long long)nNodes * IN_C, TB), TB, 0, stream>>>(summed, deg, nNodes * IN_C);
    gemm_layer<<<nb(nNodes, 16), 256, 0, stream>>>(summed, x, B1, b1, h1, nNodes);

    // 4) layer 2
    zero_f32<<<nb((long long)nNodes * HID, TB), TB, 0, stream>>>(summed, nNodes * HID);
    scatter_feats<<<nb((long long)nEdges * 32, TB), TB, 0, stream>>>(h1, src, dst, summed, nEdges);
    normalize_kernel<<<nb((long long)nNodes * HID, TB), TB, 0, stream>>>(summed, deg, nNodes * HID);
    gemm_layer<<<nb(nNodes, 16), 256, 0, stream>>>(summed, h1, B2, b2, h2, nNodes);

    // 5) global mean/max pool + head GEMM
    pool_kernel<<<nb((long long)nNodes * 32, TB), TB, 0, stream>>>(h2, batch, gsum, gmax, cnt, nNodes);
    pool_finalize<<<nb(NGR * 256, TB), TB, 0, stream>>>(gsum, gmax, cnt, gcat);
    gemm_final<<<8, 128, 0, stream>>>(gcat, BL, blin, out);
}
